// position_attention_15685220565694
// MI455X (gfx1250) — compile-verified, matrix operand order checked by reading, not run on hardware
//
#include <hip/hip_runtime.h>
#include <hip/hip_bf16.h>

typedef __attribute__((ext_vector_type(16))) _Float16 v16h;
typedef __attribute__((ext_vector_type(8)))  _Float16 v8h;
typedef __attribute__((ext_vector_type(8)))  float    v8f;

#define BB 8
#define LL 128
#define DD 256
#define NOUT 10

__device__ __forceinline__ float selu_f(float x) {
    const float SC = 1.0507009873554805f, AL = 1.6732632423543772f;
    return SC * (x >= 0.0f ? x : AL * (__expf(x) - 1.0f));
}

// ---------------------------------------------------------------------------
// Embedding gather: we[bl,d] = emb_word[word[bl],d] + emb_pos[pos[bl],d]
// ---------------------------------------------------------------------------
__global__ void embed_kernel(const int* __restrict__ word, const int* __restrict__ posn,
                             const float* __restrict__ emb_w, const float* __restrict__ emb_p,
                             float* __restrict__ we, float* __restrict__ pad) {
    int bl = blockIdx.x;
    int d  = threadIdx.x;
    int w  = word[bl];
    int p  = posn[bl];
    we[bl * DD + d] = emb_w[w * DD + d] + emb_p[p * DD + d];
    if (d == 0) pad[bl] = (w != 0) ? 1.0f : 0.0f;
}

// ---------------------------------------------------------------------------
// Weight transpose + f32->f16 convert: wt[n*K+k] = (f16) w[k*N+n]
// LDS 32x32 tile (padded stride 33 to dodge bank conflicts). grid (8,8), 256 thr.
// ---------------------------------------------------------------------------
__global__ void wcvt_kernel(const float* __restrict__ w, _Float16* __restrict__ wt) {
    __shared__ float tile[32][33];
    int k0 = blockIdx.x * 32, n0 = blockIdx.y * 32;
    int tx = threadIdx.x & 31, ty = threadIdx.x >> 5;   // 32 x 8
#pragma unroll
    for (int r = 0; r < 32; r += 8)
        tile[ty + r][tx] = w[(k0 + ty + r) * DD + (n0 + tx)];
    __syncthreads();
#pragma unroll
    for (int r = 0; r < 32; r += 8)
        wt[(n0 + ty + r) * DD + (k0 + tx)] = (_Float16)tile[tx][ty + r];
}

// ---------------------------------------------------------------------------
// WMMA GEMM: C[M,N] = epilogue(A[M,K] @ B[K,N] (+C if ACCUM) (+bias if BIAS))
// A: f32 row-major.  BT: f16, column-major (BT[n*K+k]) from wcvt_kernel.
// block = 256 threads = 8 waves; each wave one 16x16 tile; grid (M/128, N/16).
// Per-lane fragments per 32-K chunk: A = two contiguous 8-float runs (b128 x4),
// B = two contiguous 8-half runs (b128 x2). K fully unrolled -> 8 WMMAs.
// EPI: 0 none, 1 SELU, 2 ReLU
// ---------------------------------------------------------------------------
template <int EPI, bool ACCUM, bool BIAS>
__global__ void gemm_wmma(const float* __restrict__ A, const _Float16* __restrict__ BT,
                          const float* __restrict__ bias, float* __restrict__ C,
                          int M, int N, int K) {
    const int wave = threadIdx.x >> 5;
    const int lane = threadIdx.x & 31;
    const int half = lane >> 4;          // 0: lanes 0-15, 1: lanes 16-31
    const int mr   = lane & 15;
    const int m0   = blockIdx.x * 128 + wave * 16;
    const int n0   = blockIdx.y * 16;

    const float*    Arow = A  + (size_t)(m0 + mr) * K;
    const _Float16* Bcol = BT + (size_t)(n0 + mr) * K;

    v8f acc = {};
#pragma unroll
    for (int k0 = 0; k0 < DD; k0 += 32) {
        // ISA 16-bit fragment layout: element e -> K = (e<8 ? e : e+8) + 8*half
        v8f alo = *(const v8f*)(Arow + k0 + 8 * half);
        v8f ahi = *(const v8f*)(Arow + k0 + 16 + 8 * half);
        v8h blo = *(const v8h*)(Bcol + k0 + 8 * half);
        v8h bhi = *(const v8h*)(Bcol + k0 + 16 + 8 * half);
        v16h a, b;
#pragma unroll
        for (int e = 0; e < 8; ++e) {
            a[e]     = (_Float16)alo[e];
            a[e + 8] = (_Float16)ahi[e];
            b[e]     = blo[e];
            b[e + 8] = bhi[e];
        }
        acc = __builtin_amdgcn_wmma_f32_16x16x32_f16(
            /*neg_a=*/false, a, /*neg_b=*/false, b,
            /*c_mod=*/(short)0, acc, /*reuse_a=*/false, /*reuse_b=*/false);
    }

    // C/D layout: VGPR r -> M = r + 8*half ; lane -> N = mr
    const int col = n0 + mr;
    float bv = BIAS ? bias[col] : 0.0f;
#pragma unroll
    for (int r = 0; r < 8; ++r) {
        int row = m0 + r + 8 * half;
        float v = acc[r];
        if (ACCUM) v += C[(size_t)row * N + col];
        if (BIAS)  v += bv;
        if (EPI == 1) v = selu_f(v);
        if (EPI == 2) v = fmaxf(v, 0.0f);
        C[(size_t)row * N + col] = v;
    }
}

// ---------------------------------------------------------------------------
// Fused biaffine softmax-attention over keys j, independent per (b,i,d).
// Direction mask folded into loop bounds. Logits = 5*tanh(s/5) in [-5,5],
// so exp needs no max-subtraction; empty key-set rows -> 0 (matches ref).
// tanh via single v_exp: tanh(y) = (e^{2y}-1)/(e^{2y}+1), 2y = 0.4*x.
// ---------------------------------------------------------------------------
__global__ void attention_kernel(const float* __restrict__ a2, const float* __restrict__ a3,
                                 const float* __restrict__ biasv, const float* __restrict__ we1,
                                 const unsigned char* __restrict__ sl,
                                 float* __restrict__ attn, int dir) {
    int bi = blockIdx.x;                 // b*L + i
    int b  = bi >> 7;
    int i  = bi & (LL - 1);
    int d  = threadIdx.x;

    float a3v = a3[bi * DD + d] + biasv[d];
    const float* a2b  = a2  + b * LL * DD;
    const float* we1b = we1 + b * LL * DD;
    const unsigned char* slb = sl + (size_t)bi * LL;

    int jbeg = (dir == 0) ? 0       : (i + 1);
    int jend = (dir == 0) ? i       : LL;

    float denom = 0.0f, num = 0.0f;
    for (int j = jbeg; j < jend; ++j) {
        if (slb[j]) {
            float x  = a2b[j * DD + d] + a3v;
            float t  = __expf(0.4f * x);            // e^{2x/5}
            float th = (t - 1.0f) / (t + 1.0f);     // tanh(x/5)
            float e  = __expf(5.0f * th);           // exp(logit), logit in [-5,5]
            denom += e;
            num   += e * we1b[j * DD + d];
        }
    }
    attn[bi * DD + d] = (denom > 0.0f) ? (num / denom) : 0.0f;
}

// ---------------------------------------------------------------------------
// gate = sigmoid(gpre + b6 + b7 + fbias); out = (g*we1 + (1-g)*attn) * pad
// ---------------------------------------------------------------------------
__global__ void gate_kernel(const float* __restrict__ gpre, const float* __restrict__ we1,
                            const float* __restrict__ attn, const float* __restrict__ b6,
                            const float* __restrict__ b7, const float* __restrict__ fbias,
                            const float* __restrict__ pad, float* __restrict__ out) {
    int bl  = blockIdx.x;
    int d   = threadIdx.x;
    int idx = bl * DD + d;
    float g = gpre[idx] + b6[d] + b7[d] + fbias[d];
    float s = 1.0f / (1.0f + __expf(-g));
    float o = s * we1[idx] + (1.0f - s) * attn[idx];
    out[idx] = o * pad[bl];
}

// ---------------------------------------------------------------------------
// 3-way source softmax combine (pad==0 -> 0, matching softmax*pad in ref)
// ---------------------------------------------------------------------------
__global__ void combine_kernel(const float* __restrict__ of, const float* __restrict__ ob,
                               const float* __restrict__ we, const float* __restrict__ pad,
                               float* __restrict__ res) {
    int bl  = blockIdx.x;
    int idx = bl * DD + threadIdx.x;
    float p  = pad[bl];
    float v0 = of[idx], v1 = ob[idx], v2 = we[idx];
    float m  = fmaxf(v0, fmaxf(v1, v2));
    float e0 = __expf(v0 - m), e1 = __expf(v1 - m), e2 = __expf(v2 - m);
    float inv = 1.0f / (e0 + e1 + e2);
    res[idx] = p * (e0 * v0 + e1 * v1 + e2 * v2) * inv;
}

// ---------------------------------------------------------------------------
// out[b,o] = sum_{t<L*D} h[b,t] * w5[t,o] ; grid = B*NOUT blocks, LDS reduce
// ---------------------------------------------------------------------------
__global__ void final_kernel(const float* __restrict__ h, const float* __restrict__ w5,
                             float* __restrict__ out) {
    int b = blockIdx.x / NOUT;
    int o = blockIdx.x % NOUT;
    __shared__ float red[256];
    const float* hb = h + (size_t)b * (LL * DD);
    float s = 0.0f;
    for (int t = threadIdx.x; t < LL * DD; t += 256)
        s += hb[t] * w5[t * NOUT + o];
    red[threadIdx.x] = s;
    __syncthreads();
    for (int st = 128; st > 0; st >>= 1) {
        if (threadIdx.x < st) red[threadIdx.x] += red[threadIdx.x + st];
        __syncthreads();
    }
    if (threadIdx.x == 0) out[blockIdx.x] = red[0];
}

// ---------------------------------------------------------------------------
extern "C" void kernel_launch(void* const* d_in, const int* in_sizes, int n_in,
                              void* d_out, int out_size, void* d_ws, size_t ws_size,
                              hipStream_t stream) {
    const int*           word  = (const int*)d_in[0];
    const int*           posn  = (const int*)d_in[1];
    const unsigned char* sl    = (const unsigned char*)d_in[2];   // bool [B,L,L]
    const float*         emb_w = (const float*)d_in[3];
    const float*         emb_p = (const float*)d_in[4];
    const float*         w_d4  = (const float*)d_in[25];
    const float*         b_d4  = (const float*)d_in[26];
    const float*         w_d5  = (const float*)d_in[27];

    const int NBL = BB * LL;            // 1024
    const size_t SZ = (size_t)NBL * DD; // 262144 floats per activation buffer
    const size_t WSZ = (size_t)DD * DD; // 65536 elems per weight

    float* ws   = (float*)d_ws;
    float* we   = ws + 0 * SZ;
    float* we1  = ws + 1 * SZ;
    float* a2   = ws + 2 * SZ;
    float* a3   = ws + 3 * SZ;
    float* attn = ws + 4 * SZ;
    float* gpre = ws + 5 * SZ;
    float* outf = ws + 6 * SZ;
    float* outb = ws + 7 * SZ;
    float* comb = ws + 8 * SZ;
    float* hbuf = ws + 9 * SZ;
    float* pad  = ws + 10 * SZ;         // NBL floats
    // f16 transposed weights: 11 slots of 256x256 halves (16B-aligned offset)
    _Float16* wt = (_Float16*)(ws + 10 * SZ + NBL);

    dim3 gemm_grid(NBL / 128, DD / 16); // (8, 16)
    dim3 wgrid(DD / 32, DD / 32);       // (8, 8)

    // Pre-convert all 11 square weights: f32 row-major -> f16 column-major.
    // slots: dir*5 + {w1,w2,w3,w6,w7}, slot 10 = w_d4
    for (int dir = 0; dir < 2; ++dir) {
        int base = 5 + dir * 10;
        const int widx[5] = {base + 0, base + 2, base + 3, base + 6, base + 8};
        for (int t = 0; t < 5; ++t)
            wcvt_kernel<<<wgrid, 256, 0, stream>>>((const float*)d_in[widx[t]],
                                                   wt + (size_t)(dir * 5 + t) * WSZ);
    }
    wcvt_kernel<<<wgrid, 256, 0, stream>>>(w_d4, wt + 10 * WSZ);

    embed_kernel<<<NBL, DD, 0, stream>>>(word, posn, emb_w, emb_p, we, pad);

    for (int dir = 0; dir < 2; ++dir) {
        int base = 5 + dir * 10;
        const float* b1    = (const float*)d_in[base + 1];
        const float* biasv = (const float*)d_in[base + 4];
        const float* fbias = (const float*)d_in[base + 5];
        const float* b6    = (const float*)d_in[base + 7];
        const float* b7    = (const float*)d_in[base + 9];
        const _Float16* t1 = wt + (size_t)(dir * 5 + 0) * WSZ;
        const _Float16* t2 = wt + (size_t)(dir * 5 + 1) * WSZ;
        const _Float16* t3 = wt + (size_t)(dir * 5 + 2) * WSZ;
        const _Float16* t6 = wt + (size_t)(dir * 5 + 3) * WSZ;
        const _Float16* t7 = wt + (size_t)(dir * 5 + 4) * WSZ;
        float* odir = (dir == 0) ? outf : outb;

        gemm_wmma<1, false, true ><<<gemm_grid, 256, 0, stream>>>(we,  t1, b1,      we1,  NBL, DD, DD);
        gemm_wmma<0, false, false><<<gemm_grid, 256, 0, stream>>>(we1, t2, nullptr, a2,   NBL, DD, DD);
        gemm_wmma<0, false, false><<<gemm_grid, 256, 0, stream>>>(we1, t3, nullptr, a3,   NBL, DD, DD);
        attention_kernel<<<NBL, DD, 0, stream>>>(a2, a3, biasv, we1, sl, attn, dir);
        gemm_wmma<0, false, false><<<gemm_grid, 256, 0, stream>>>(we1,  t6, nullptr, gpre, NBL, DD, DD);
        gemm_wmma<0, true,  false><<<gemm_grid, 256, 0, stream>>>(attn, t7, nullptr, gpre, NBL, DD, DD);
        gate_kernel<<<NBL, DD, 0, stream>>>(gpre, we1, attn, b6, b7, fbias, pad, odir);
    }

    combine_kernel<<<NBL, DD, 0, stream>>>(outf, outb, we, pad, comb);
    gemm_wmma<2, false, true><<<gemm_grid, 256, 0, stream>>>(comb, wt + 10 * WSZ, b_d4, hbuf, NBL, DD, DD);
    final_kernel<<<BB * NOUT, 256, 0, stream>>>(hbuf, w_d5, (float*)d_out);
}
